// DeepFM_27169963114981
// MI455X (gfx1250) — compile-verified, matrix operand order checked by reading, not run on hardware
//
#include <hip/hip_runtime.h>
#include <hip/hip_bf16.h>

typedef __bf16 bf16_t;
typedef __attribute__((ext_vector_type(16))) __bf16 v16bf;
typedef __attribute__((ext_vector_type(8)))  __bf16 v8bf;
typedef __attribute__((ext_vector_type(8)))  float  v8f;

// ---- problem constants ----
constexpr int BATCH = 16384;
constexpr int FD    = 13;
constexpr int NF    = 39;     // 13 dense + 26 sparse fields
constexpr int EMB   = 16;
constexpr int VOCAB = 100000;
constexpr int DIN   = 624;    // 39*16
constexpr int K1    = 640;    // DIN padded to multiple of 32
constexpr int H1    = 512;
constexpr int H2    = 256;
constexpr int ROWS  = 64;     // rows per block
constexpr float INVEPS = 0.99999500003749969f; // 1/sqrt(1+1e-5)

// ---- ws layout (elements of bf16 for packs) ----
constexpr int KT1 = K1 / 32;          // 20
constexpr int NT1 = H1 / 16;          // 32
constexpr int KT2 = H1 / 32;          // 16
constexpr int NT2 = H2 / 16;          // 16
constexpr int PACK1_ELEMS = KT1 * NT1 * 32 * 16;  // 327680
constexpr int PACK2_ELEMS = KT2 * NT2 * 32 * 16;  // 131072
constexpr size_t CONST_OFF_BYTES = (size_t)(PACK1_ELEMS + PACK2_ELEMS) * 2; // 917504

// Per-lane K sequence for 16-bit WMMA A/B operands (ISA 7.12.2):
// lanes 0-15 hold K = {0..7,16..23}; lanes 16-31 hold K = {8..15,24..31}
__device__ __host__ inline int kseq(int hl, int j) {
    return j + 8 * hl + ((j >= 8) ? 8 : 0);
}

// ---- pack Lw1 (624x512 fp32, row-major KxN) into WMMA-B bf16 tiles, K padded ----
__global__ void pack_lw1(const float* __restrict__ Lw1, bf16_t* __restrict__ pack) {
    int tid = blockIdx.x * 256 + threadIdx.x;
    if (tid >= PACK1_ELEMS) return;
    int j    = tid & 15;
    int lane = (tid >> 4) & 31;
    int nt   = (tid >> 9) & 31;
    int kt   = tid >> 14;
    int k = kt * 32 + kseq(lane >> 4, j);
    int n = nt * 16 + (lane & 15);
    float v = (k < DIN) ? Lw1[k * H1 + n] : 0.0f;
    pack[tid] = (bf16_t)v;
}

// ---- pack Lw2 (512x256 fp32) ----
__global__ void pack_lw2(const float* __restrict__ Lw2, bf16_t* __restrict__ pack) {
    int tid = blockIdx.x * 256 + threadIdx.x;
    if (tid >= PACK2_ELEMS) return;
    int j    = tid & 15;
    int lane = (tid >> 4) & 31;
    int nt   = (tid >> 9) & 15;
    int kt   = tid >> 13;
    int k = kt * 32 + kseq(lane >> 4, j);
    int n = nt * 16 + (lane & 15);
    pack[tid] = (bf16_t)Lw2[k * H2 + n];
}

// ---- const2 = sum_n (Lb2[n]*inv*g2[n] + be2[n]) ----
__global__ void prep_consts(const float* __restrict__ Lb2, const float* __restrict__ g2,
                            const float* __restrict__ be2, float* __restrict__ cptr) {
    __shared__ float red[256];
    int t = threadIdx.x; // H2 == 256 == blockDim
    red[t] = Lb2[t] * (INVEPS * g2[t]) + be2[t];
    __syncthreads();
    for (int s = 128; s > 0; s >>= 1) {
        if (t < s) red[t] += red[t + s];
        __syncthreads();
    }
    if (t == 0) *cptr = red[0];
}

// ---- fused main kernel: gathers + FM + bf16 WMMA MLP, all in LDS ----
__global__ __launch_bounds__(256) void deepfm_main(
    const int* __restrict__ Xi, const float* __restrict__ Xv,
    const float* __restrict__ W1, const float* __restrict__ b1, const float* __restrict__ emb1,
    const float* __restrict__ W2, const float* __restrict__ b2, const float* __restrict__ emb2,
    const float* __restrict__ Lb1, const float* __restrict__ g1, const float* __restrict__ be1,
    const float* __restrict__ g2, const float* __restrict__ bias,
    const bf16_t* __restrict__ pack1, const bf16_t* __restrict__ pack2,
    const float* __restrict__ cptr, float* __restrict__ out)
{
    __shared__ bf16_t hbuf[ROWS * K1];   // 80 KB : h tile (bf16, K-padded)
    __shared__ bf16_t y1b [ROWS * H1];   // 64 KB : layer-1 activations (bf16)
    __shared__ float  fmp [ROWS];        // FM part + bias per row
    __shared__ float  racc[ROWS];        // layer-2 row sums

    const int tid  = threadIdx.x;
    const int lane = tid & 31;
    const int wid  = tid >> 5;
    const int hl   = lane >> 4;     // half of wave (selects row / K-subset)
    const int e    = lane & 15;     // embedding dim / N-within-tile
    const int b0   = blockIdx.x * ROWS;

    // ---------- Phase 1: gathers, FM first+second order, build h tile ----------
    for (int p = 0; p < 4; p++) {
        const int row = p * 16 + wid * 2 + hl;
        const int b   = b0 + row;
        const int*   xi_p = Xi + b * NF;
        const float* xv_p = Xv + b * NF;
        float f1 = 0.f, s = 0.f, sq = 0.f;
        #pragma unroll
        for (int f = 0; f < FD; f++) {
            float xi = (float)xi_p[f];
            float xv = xv_p[f];
            f1 += (xi * W1[f * EMB + e] + b1[f * EMB + e]) * xv;
            float sec = xi * W2[f * EMB + e] + b2[f * EMB + e];
            s += sec; sq += sec * sec;
            hbuf[row * K1 + f * EMB + e] = (bf16_t)sec;
        }
        for (int f = FD; f < NF; f++) {
            int t   = f - FD;
            int idx = xi_p[f];
            float xv = xv_p[f];
            int base = (t * VOCAB + idx) * EMB + e;
            f1 += emb1[base] * xv;
            float sec = emb2[base] * xv;
            s += sec; sq += sec * sec;
            hbuf[row * K1 + f * EMB + e] = (bf16_t)sec;
        }
        hbuf[row * K1 + DIN + e] = (bf16_t)0.0f;  // K padding
        float v = f1 + 0.5f * (s * s - sq);
        for (int m = 1; m < 16; m <<= 1) v += __shfl_xor(v, m, 32);
        if (e == 0) { fmp[row] = v + bias[b]; racc[row] = 0.f; }
    }
    __syncthreads();

    const v8f vzero = {0.f, 0.f, 0.f, 0.f, 0.f, 0.f, 0.f, 0.f};

    // ---------- Phase 2: y1 = (h @ Lw1 + Lb1)*inv*g1 + be1 ----------
    for (int nt = wid; nt < NT1; nt += 8) {
        const int n = nt * 16 + e;
        const float c1 = INVEPS * g1[n];
        const float lb = Lb1[n];
        const float bv = be1[n];
        v8f acc[4];
        #pragma unroll
        for (int mt = 0; mt < 4; mt++) acc[mt] = vzero;
        // Keep rolled: full unroll spills A-fragments to scratch (seen in round 1).
        #pragma unroll 1
        for (int kt = 0; kt < KT1; kt++) {
            v16bf Bv = *(const v16bf*)(pack1 + ((size_t)(kt * NT1 + nt) * 32 + lane) * 16);
            #pragma unroll
            for (int mt = 0; mt < 4; mt++) {
                const bf16_t* ap = hbuf + (mt * 16 + e) * K1 + kt * 32;
                union { v16bf v; v8bf h[2]; } A;
                A.h[0] = *(const v8bf*)(ap + 8 * hl);
                A.h[1] = *(const v8bf*)(ap + 16 + 8 * hl);
                acc[mt] = __builtin_amdgcn_wmma_f32_16x16x32_bf16(
                    false, A.v, false, Bv, (short)0, acc[mt], false, false);
            }
        }
        #pragma unroll
        for (int mt = 0; mt < 4; mt++) {
            #pragma unroll
            for (int r = 0; r < 8; r++) {
                int m = mt * 16 + r + 8 * hl;     // C/D layout: vgpr r, lane half
                float yv = (acc[mt][r] + lb) * c1 + bv;
                y1b[m * H1 + n] = (bf16_t)yv;
            }
        }
    }
    __syncthreads();

    // ---------- Phase 3: rowsum of (y1 @ Lw2)*inv*g2  ----------
    for (int nt = wid; nt < NT2; nt += 8) {
        const int n = nt * 16 + e;
        const float c2 = INVEPS * g2[n];
        v8f acc[4];
        #pragma unroll
        for (int mt = 0; mt < 4; mt++) acc[mt] = vzero;
        #pragma unroll 1
        for (int kt = 0; kt < KT2; kt++) {
            v16bf Bv = *(const v16bf*)(pack2 + ((size_t)(kt * NT2 + nt) * 32 + lane) * 16);
            #pragma unroll
            for (int mt = 0; mt < 4; mt++) {
                const bf16_t* ap = y1b + (mt * 16 + e) * H1 + kt * 32;
                union { v16bf v; v8bf h[2]; } A;
                A.h[0] = *(const v8bf*)(ap + 8 * hl);
                A.h[1] = *(const v8bf*)(ap + 16 + 8 * hl);
                acc[mt] = __builtin_amdgcn_wmma_f32_16x16x32_bf16(
                    false, A.v, false, Bv, (short)0, acc[mt], false, false);
            }
        }
        #pragma unroll
        for (int mt = 0; mt < 4; mt++) {
            #pragma unroll
            for (int r = 0; r < 8; r++) {
                atomicAdd(&racc[mt * 16 + r + 8 * hl], acc[mt][r] * c2);
            }
        }
    }
    __syncthreads();

    if (tid < ROWS) out[b0 + tid] = fmp[tid] + racc[tid] + *cptr;
}

extern "C" void kernel_launch(void* const* d_in, const int* in_sizes, int n_in,
                              void* d_out, int out_size, void* d_ws, size_t ws_size,
                              hipStream_t stream) {
    const int*   Xi   = (const int*)  d_in[0];
    const float* Xv   = (const float*)d_in[1];
    const float* W1   = (const float*)d_in[2];
    const float* b1   = (const float*)d_in[3];
    const float* emb1 = (const float*)d_in[4];
    const float* W2   = (const float*)d_in[5];
    const float* b2   = (const float*)d_in[6];
    const float* emb2 = (const float*)d_in[7];
    const float* Lw1  = (const float*)d_in[8];
    const float* Lb1  = (const float*)d_in[9];
    const float* g1   = (const float*)d_in[10];
    const float* be1  = (const float*)d_in[11];
    const float* Lw2  = (const float*)d_in[12];
    const float* Lb2  = (const float*)d_in[13];
    const float* g2   = (const float*)d_in[14];
    const float* be2  = (const float*)d_in[15];
    const float* bias = (const float*)d_in[16];

    bf16_t* pack1 = (bf16_t*)d_ws;
    bf16_t* pack2 = pack1 + PACK1_ELEMS;
    float*  cptr  = (float*)((char*)d_ws + CONST_OFF_BYTES);

    pack_lw1<<<(PACK1_ELEMS + 255) / 256, 256, 0, stream>>>(Lw1, pack1);
    pack_lw2<<<(PACK2_ELEMS + 255) / 256, 256, 0, stream>>>(Lw2, pack2);
    prep_consts<<<1, 256, 0, stream>>>(Lb2, g2, be2, cptr);

    deepfm_main<<<BATCH / ROWS, 256, 0, stream>>>(
        Xi, Xv, W1, b1, emb1, W2, b2, emb2,
        Lb1, g1, be1, g2, bias, pack1, pack2, cptr, (float*)d_out);
}
